// DeformConvLayer_40329742910044
// MI455X (gfx1250) — matrix operand order repaired and checked by reading, not compiled, not run on hardware
//
#include <hip/hip_runtime.h>

typedef __attribute__((ext_vector_type(16))) _Float16 v16h;
typedef __attribute__((ext_vector_type(8)))  float    v8f;
typedef __attribute__((ext_vector_type(4)))  float    v4f;
typedef __attribute__((ext_vector_type(4)))  unsigned int v4u;

#define NB   4
#define CIN  256
#define COUT 256
#define HH   96
#define WW   96
#define HW   (HH * WW)   /* 9216 */
#define NOFF 18
#define NPAD 32          /* offset-conv N padded to 2 WMMA tiles */
#define TM   128         /* pixels per workgroup tile            */
#define KC   32          /* K chunk (WMMA f16 K)                 */

// ws layout (bytes): [offs f32: 4*18*9216*4 = 2654208][Bpack f16: 9*256*256*2
// = 1179648][Opack f16: 9*32*256*2 = 147456]  -- all 256B aligned
#define WS_OFFS_BYTES  (NB * NOFF * HW * 4)
#define WS_BPACK_BYTES (9 * COUT * CIN * 2)

// ---------------------------------------------------------------------------
// Kernel 0: one-shot weight repack to f16.
//   Bpack[k2][oc][c] = w_def[oc][c][k2]
//   Opack[k2][o][c]  = (o<18) ? w_off[o][c][k2] : 0      (N padded to 32)
// Makes every B-tile stage in the GEMM kernels a contiguous per-thread load (8B in kernel 1, 64B in kernel 2).
// ---------------------------------------------------------------------------
__global__ __launch_bounds__(256) void pack_weights_kernel(
    const float* __restrict__ w_def, const float* __restrict__ w_off,
    _Float16* __restrict__ Bpack, _Float16* __restrict__ Opack)
{
  int idx = blockIdx.x * 256 + threadIdx.x;
  if (idx < 9 * COUT * CIN) {
    int c  = idx % CIN;
    int oc = (idx / CIN) % COUT;
    int k2 = idx / (CIN * COUT);
    Bpack[idx] = (_Float16)w_def[(size_t)oc * (CIN * 9) + c * 9 + k2];
  }
  if (idx < 9 * NPAD * CIN) {
    int c  = idx % CIN;
    int o  = (idx / CIN) % NPAD;
    int k2 = idx / (CIN * NPAD);
    float v = (o < NOFF) ? w_off[(size_t)o * (CIN * 9) + c * 9 + k2] : 0.0f;
    Opack[idx] = (_Float16)v;
  }
}

// ---------------------------------------------------------------------------
// Kernel 1: offset conv as implicit GEMM (M=36864, N=32(pad18), K=2304) with
// v_wmma_f32_16x16x32_f16. A-tile = shifted-window im2col gather of x.
// ---------------------------------------------------------------------------
__global__ __launch_bounds__(256) void offset_wmma_kernel(
    const float* __restrict__ x, const _Float16* __restrict__ Opack,
    const float* __restrict__ b_off, float* __restrict__ offs)
{
  __shared__ __align__(16) _Float16 Alds[TM * KC];
  __shared__ __align__(16) _Float16 Blds[NPAD * KC];

  const int tid  = threadIdx.x;
  const int wave = tid >> 5;
  const int lane = tid & 31;
  const int p0   = blockIdx.x * TM;      // HW % TM == 0 -> tile stays in one b
  const int b    = p0 / HW;
  const int hw0  = p0 % HW;

  v8f acc[2] = {};
  const float* xb = x + (size_t)b * CIN * HW;

  // A-fill thread role: pixel m, K half k0
  const int m  = tid & (TM - 1);
  const int k0 = (tid >> 7) * 16;        // 0 or 16
  const int hw = hw0 + m;
  const int ho = hw / WW, wo = hw % WW;

  for (int k2 = 0; k2 < 9; ++k2) {
    int y  = ho - 1 + (k2 / 3);
    int xx = wo - 1 + (k2 % 3);
    bool valid = ((unsigned)y < (unsigned)HH) && ((unsigned)xx < (unsigned)WW);
    int  sidx  = valid ? (y * WW + xx) : 0;          // always in-bounds
    float vmask = valid ? 1.0f : 0.0f;

    for (int c0 = 0; c0 < CIN; c0 += KC) {
      // ---- A tile: 128 x 32, shifted-window gather ----
      {
        const float* xc = xb + (size_t)(c0 + k0) * HW + sidx;
        _Float16 tmp[16];
        #pragma unroll
        for (int j = 0; j < 16; ++j) {
          tmp[j] = (_Float16)(vmask * xc[0]);
          xc += HW;
        }
        *(v4u*)&Alds[m * KC + k0]     = *(v4u*)&tmp[0];
        *(v4u*)&Alds[m * KC + k0 + 8] = *(v4u*)&tmp[8];
      }
      // ---- B tile: 32 x 32 from packed f16, contiguous ----
      {
        int n   = tid & (NPAD - 1);
        int kk0 = (tid >> 5) * 4;                    // 0..28
        const _Float16* op = Opack + ((size_t)k2 * NPAD + n) * CIN + c0 + kk0;
        #pragma unroll
        for (int j = 0; j < 4; ++j) Blds[n * KC + kk0 + j] = op[j];
      }
      __syncthreads();

      union { v16h v; v4u u[2]; } Af, Bf;
      {
        int mm = wave * 16 + (lane & 15);
        int ka = (lane < 16) ? 0 : 8;
        Af.u[0] = *(const v4u*)&Alds[mm * KC + ka];
        Af.u[1] = *(const v4u*)&Alds[mm * KC + ka + 16];
      }
      #pragma unroll
      for (int j = 0; j < 2; ++j) {
        int n  = j * 16 + (lane & 15);
        int kb = (lane < 16) ? 0 : 8;
        Bf.u[0] = *(const v4u*)&Blds[n * KC + kb];
        Bf.u[1] = *(const v4u*)&Blds[n * KC + kb + 16];
        acc[j] = __builtin_amdgcn_wmma_f32_16x16x32_f16(
            false, Af.v, false, Bf.v, (short)0, acc[j], false, false);
      }
      __syncthreads();
    }
  }

  // ---- epilogue: only real 18 offset channels are written ----
  {
    int nlo  = lane & 15;
    int mofs = (lane < 16) ? 0 : 8;
    #pragma unroll
    for (int j = 0; j < 2; ++j) {
      int oc = j * 16 + nlo;
      if (oc < NOFF) {
        float bd = b_off[oc];
        float* op = offs + (size_t)(b * NOFF + oc) * HW + hw0 + wave * 16 + mofs;
        v4f lo = { acc[j][0] + bd, acc[j][1] + bd, acc[j][2] + bd, acc[j][3] + bd };
        v4f hi = { acc[j][4] + bd, acc[j][5] + bd, acc[j][6] + bd, acc[j][7] + bd };
        *(v4f*)(op)     = lo;
        *(v4f*)(op + 4) = hi;
      }
    }
  }
}

// ---------------------------------------------------------------------------
// Kernel 2: fused bilinear sampling + implicit GEMM, full N=256 in one pass
// so the dominant gather/interp A-fill work is done exactly once.
// Tile: M=128 x N=256, 8 waves (one 16-row strip each, 16 WMMAs per chunk).
// ---------------------------------------------------------------------------
__global__ __launch_bounds__(256) void deform_wmma_kernel(
    const float* __restrict__ x, const float* __restrict__ offs,
    const _Float16* __restrict__ Bpack, const float* __restrict__ b_def,
    float* __restrict__ out)
{
  __shared__ __align__(16) _Float16 Alds[TM * KC];     // 8 KB
  __shared__ __align__(16) _Float16 Blds[COUT * KC];   // 16 KB
  __shared__ __align__(16) float    swt[TM * 4];
  __shared__ __align__(16) int      sidx[TM * 4];

  const int tid  = threadIdx.x;
  const int wave = tid >> 5;
  const int lane = tid & 31;
  const int p0   = blockIdx.x * TM;
  const int b    = p0 / HW;
  const int hw0  = p0 % HW;

  v8f acc[16] = {};                      // 16 N sub-tiles of 16
  const float* xb = x + (size_t)b * CIN * HW;

  for (int k2 = 0; k2 < 9; ++k2) {
    // ---- per-pixel bilinear metadata (threads 0..127) ----
    if (tid < TM) {
      int hw = hw0 + tid;
      int ho = hw / WW, wo = hw % WW;
      float oy = offs[(size_t)(b * NOFF + 2 * k2    ) * HW + hw];
      float ox = offs[(size_t)(b * NOFF + 2 * k2 + 1) * HW + hw];
      float py = (float)(ho - 1 + (k2 / 3)) + oy;
      float px = (float)(wo - 1 + (k2 % 3)) + ox;
      float y0f = floorf(py), x0f = floorf(px);
      float wy1 = py - y0f, wy0 = 1.0f - wy1;
      float wx1 = px - x0f, wx0 = 1.0f - wx1;
      int y0 = (int)y0f, x0 = (int)x0f;
      #pragma unroll
      for (int cI = 0; cI < 4; ++cI) {
        int yi = y0 + (cI >> 1);
        int xi = x0 + (cI & 1);
        float wv = ((cI >> 1) ? wy1 : wy0) * ((cI & 1) ? wx1 : wx0);
        bool valid = (yi >= 0) && (yi < HH) && (xi >= 0) && (xi < WW);
        int yc = min(max(yi, 0), HH - 1);
        int xc = min(max(xi, 0), WW - 1);
        swt [tid * 4 + cI] = valid ? wv : 0.0f;
        sidx[tid * 4 + cI] = yc * WW + xc;
      }
    }
    __syncthreads();

    for (int c0 = 0; c0 < CIN; c0 += KC) {
      // ---- A tile: sample 128 pixels x 32 channels ----
      {
        int m  = tid & (TM - 1);
        int k0 = (tid >> 7) * 16;
        float w0 = swt[m * 4 + 0], w1 = swt[m * 4 + 1];
        float w2 = swt[m * 4 + 2], w3 = swt[m * 4 + 3];
        int   i0 = sidx[m * 4 + 0], i1 = sidx[m * 4 + 1];
        int   i2 = sidx[m * 4 + 2], i3 = sidx[m * 4 + 3];
        const float* xc = xb + (size_t)(c0 + k0) * HW;
        _Float16 tmp[16];
        #pragma unroll
        for (int j = 0; j < 16; ++j) {
          float v = w0 * xc[i0] + w1 * xc[i1] + w2 * xc[i2] + w3 * xc[i3];
          tmp[j] = (_Float16)v;
          xc += HW;
        }
        *(v4u*)&Alds[m * KC + k0]     = *(v4u*)&tmp[0];
        *(v4u*)&Alds[m * KC + k0 + 8] = *(v4u*)&tmp[8];
      }
      // ---- B tile: 256 x 32 from packed f16, 64 contiguous bytes/thread ----
      {
        const _Float16* bp = Bpack + ((size_t)k2 * COUT + tid) * CIN + c0;
        v4u* dst = (v4u*)&Blds[tid * KC];
        const v4u* src = (const v4u*)bp;
        dst[0] = src[0]; dst[1] = src[1]; dst[2] = src[2]; dst[3] = src[3];
      }
      __syncthreads();

      // ---- WMMA: one A fragment x 16 B fragments ----
      union { v16h v; v4u u[2]; } Af, Bf;
      {
        int mm = wave * 16 + (lane & 15);
        int ka = (lane < 16) ? 0 : 8;
        Af.u[0] = *(const v4u*)&Alds[mm * KC + ka];
        Af.u[1] = *(const v4u*)&Alds[mm * KC + ka + 16];
      }
      #pragma unroll
      for (int j = 0; j < 16; ++j) {
        int n  = j * 16 + (lane & 15);
        int kb = (lane < 16) ? 0 : 8;
        Bf.u[0] = *(const v4u*)&Blds[n * KC + kb];
        Bf.u[1] = *(const v4u*)&Blds[n * KC + kb + 16];
        acc[j] = __builtin_amdgcn_wmma_f32_16x16x32_f16(
            false, Af.v, false, Bf.v, (short)0, acc[j], false, false);
      }
      __syncthreads();
    }
  }

  // ---- epilogue: contiguous 8-float strip per lane per sub-tile, + bias ----
  {
    int nlo  = lane & 15;
    int mofs = (lane < 16) ? 0 : 8;
    #pragma unroll
    for (int j = 0; j < 16; ++j) {
      int oc = j * 16 + nlo;
      float bd = b_def[oc];
      float* op = out + (size_t)(b * COUT + oc) * HW + hw0 + wave * 16 + mofs;
      v4f lo = { acc[j][0] + bd, acc[j][1] + bd, acc[j][2] + bd, acc[j][3] + bd };
      v4f hi = { acc[j][4] + bd, acc[j][5] + bd, acc[j][6] + bd, acc[j][7] + bd };
      *(v4f*)(op)     = lo;
      *(v4f*)(op + 4) = hi;
    }
  }
}

// ---------------------------------------------------------------------------
extern "C" void kernel_launch(void* const* d_in, const int* in_sizes, int n_in,
                              void* d_out, int out_size, void* d_ws, size_t ws_size,
                              hipStream_t stream)
{
  const float* x     = (const float*)d_in[0];
  const float* w_off = (const float*)d_in[1];
  const float* b_off = (const float*)d_in[2];
  const float* w_def = (const float*)d_in[3];
  const float* b_def = (const float*)d_in[4];
  float* out = (float*)d_out;

  float*    offs  = (float*)d_ws;
  _Float16* Bpack = (_Float16*)((char*)d_ws + WS_OFFS_BYTES);
  _Float16* Opack = (_Float16*)((char*)d_ws + WS_OFFS_BYTES + WS_BPACK_BYTES);

  (void)in_sizes; (void)n_in; (void)out_size; (void)ws_size;

  pack_weights_kernel<<<(9 * COUT * CIN) / 256, 256, 0, stream>>>(
      w_def, w_off, Bpack, Opack);

  const int ntiles = NB * HW / TM;            // 288
  offset_wmma_kernel<<<ntiles, 256, 0, stream>>>(x, Opack, b_off, offs);
  deform_wmma_kernel<<<ntiles, 256, 0, stream>>>(x, offs, Bpack, b_def, out);
}
